// UltraAdvancedModel_37967510896682
// MI455X (gfx1250) — compile-verified, matrix operand order checked by reading, not run on hardware
//
// MI455X (gfx1250, CDNA5) implementation of the LSTM+attention location model.
//
// Strategy: bf16 WMMA (v_wmma_f32_16x16x32_bf16, f32 accum) for every GEMM;
// the whole weight set fits in the 192MB L2 after a one-time bf16 convert, so
// weight tiles stream from global (L2-hit) per wave. All weights are stored
// TRANSPOSED (N x K) so both WMMA operands are two b128 loads per lane
// (ISA wave32 layout: lane needs K = half*8+{0..7} and 16+half*8+{0..7}).
// GEMM wave tile = 16x32 (2 accumulators sharing one A fragment): this is the
// largest tile that compiles spill-free; the compiler clause-batches the b128
// loads and unrolls the K loop itself (verified in codegen).
// LSTM recurrence = one persistent 32-wave workgroup, h/c/gates in LDS.
// Attention = online-softmax (flash), hd=32 == one WMMA K-step; V stored
// transposed per head; V loads hoisted above the softmax VALU to overlap.
// Softmax/sigmoid/tanh use __expf (v_exp_f32 TRANS op, co-executes with XDL);
// GELU keeps exact erff to match approximate=False.

#include <hip/hip_runtime.h>
#include <math.h>

#define B_ 16
#define S_ 512
#define D_ 256
#define NH_ 8
#define HD_ 32
#define FF_ 1024
#define L_ 4
#define NLOC_ 40000
#define NC_ 64
#define BS_ (B_ * S_)   // 8192

typedef __attribute__((ext_vector_type(16))) __bf16 v16bf;
typedef __attribute__((ext_vector_type(8)))  float  v8f;

#define DEV __device__ __forceinline__

DEV v8f wmma_bf16(v16bf a, v16bf b, v8f c) {
  return __builtin_amdgcn_wmma_f32_16x16x32_bf16(false, a, false, b, (short)0, c,
                                                 false, false);
}

// Two b128 loads -> one 16x32-bf16 operand fragment (global or LDS generic
// pointer; p0 must be 16B aligned; covers K offsets 0..7 and 16..23 relative
// to p0 == base + k0 + half*8).
DEV v16bf ld_frag16(const __bf16* p0) {
  union { uint4 q[2]; v16bf v; } u;
  const uint4* q = (const uint4*)p0;
  u.q[0] = q[0];
  u.q[1] = q[2];   // +32 bytes == K offset +16
  return u.v;
}

// A fragment: rows row0..row0+15 (clamped to M), cols k0..k0+31, row-major lda.
DEV v16bf load_a(const __bf16* A, int lda, int row0, int k0, int M) {
  int lane = threadIdx.x & 31, half = lane >> 4, m = lane & 15;
  int r = row0 + m; if (r >= M) r = M - 1;
  return ld_frag16(A + (size_t)r * lda + k0 + half * 8);
}

// B fragment from K-contiguous storage: B[k][n] = src[(n0+n)*ldn + k].
// Used for transposed weights (N x K), K^T in attention, and V^T.
DEV v16bf load_bT(const __bf16* src, int ldn, int n0) {
  int lane = threadIdx.x & 31, half = lane >> 4, n = lane & 15;
  return ld_frag16(src + (size_t)(n0 + n) * ldn + half * 8);
}

DEV float sigf(float x) { return 1.f / (1.f + __expf(-x)); }
DEV float tanh_fast(float x) { return 1.f - 2.f / (__expf(2.f * x) + 1.f); }
DEV float geluf(float x) { return 0.5f * x * (1.f + erff(x * 0.70710678118654752f)); }

// ---------------------------------------------------------------------------
// Generic GEMM: C[M,N] = act(A_bf16[M,K] @ W[K,N] + bias) with W stored
// transposed (WT: N x K). f32 out plus optional strided bf16 mirror.
// Block = 128 thr (4 waves), block tile 32x64, wave tile 16x32 (2 C frags
// sharing one A frag -> 2 independent WMMAs per K step). N % 64 == 0.
// M handled with clamped loads / guarded stores. act: 0 none, 1 sigmoid,
// 2 exact gelu.
// ---------------------------------------------------------------------------
__global__ void gemm_bf16_kernel(const __bf16* __restrict__ A, int lda,
                                 const __bf16* __restrict__ WT,
                                 const float* __restrict__ bias,
                                 float* __restrict__ C, int ldc,
                                 __bf16* __restrict__ Cb, int ldcb,
                                 int M, int N, int K, int act) {
  int wid = threadIdx.x >> 5;
  int mt = wid >> 1, nt = wid & 1;
  int row0 = blockIdx.y * 32 + mt * 16;
  int col0 = blockIdx.x * 64 + nt * 32;
  int lane = threadIdx.x & 31, half = lane >> 4, n = lane & 15;
  v8f acc0 = {}, acc1 = {};
  for (int k0 = 0; k0 < K; k0 += 32) {
    if (k0 + 32 < K)  // global_prefetch_b8 of next A tile
      __builtin_prefetch(A + (size_t)row0 * lda + k0 + 32, 0, 1);
    v16bf a = load_a(A, lda, row0, k0, M);
    v16bf b0 = load_bT(WT + k0, K, col0);
    v16bf b1 = load_bT(WT + k0, K, col0 + 16);
    acc0 = wmma_bf16(a, b0, acc0);
    acc1 = wmma_bf16(a, b1, acc1);
  }
#pragma unroll
  for (int t = 0; t < 2; ++t) {
    v8f acc = t ? acc1 : acc0;
    int col = col0 + t * 16 + n;
    float bv = bias ? bias[col] : 0.f;
#pragma unroll
    for (int r = 0; r < 8; ++r) {
      float v = acc[r] + bv;
      if (act == 1) v = sigf(v);
      else if (act == 2) v = geluf(v);
      int row = row0 + r + 8 * half;
      if (row < M) {
        C[(size_t)row * ldc + col] = v;
        if (Cb) Cb[(size_t)row * ldcb + col] = (__bf16)v;
      }
    }
  }
}

// ---------------------------------------------------------------------------
// LSTM recurrence: one persistent workgroup (1024 thr = 32 waves).
// h (16x256 bf16), c (16x256 f32), gate pre-acts (16x1024 f32) in LDS (88KB).
// Wave tile = 16x32 of h@Whh (A frag shared across both N-tiles). WhhT: N x K.
// ---------------------------------------------------------------------------
__global__ void lstm_rec_kernel(const float* __restrict__ xg,
                                const __bf16* __restrict__ WhhT,
                                float* __restrict__ lo) {
  __shared__ __align__(16) __bf16 hls[16 * 256];
  __shared__ float cls[16 * 256];
  __shared__ float gls[16 * 1024];
  int tid = threadIdx.x;
  int wid = tid >> 5, lane = tid & 31;
  int half = lane >> 4, n = lane & 15;
  int n0 = wid * 32;
  for (int i = tid; i < 16 * 256; i += 1024) { hls[i] = (__bf16)0.f; cls[i] = 0.f; }
  __syncthreads();
  for (int t = 0; t < S_; ++t) {
    v8f acc0, acc1;
#pragma unroll
    for (int r = 0; r < 8; ++r) {
      int b = r + 8 * half;
      const float* xr = xg + ((size_t)(b * S_ + t)) * 1024 + n0 + n;
      acc0[r] = xr[0];
      acc1[r] = xr[16];
    }
    for (int k0 = 0; k0 < 256; k0 += 32) {
      v16bf a = load_a(hls, 256, 0, k0, 16);
      v16bf w0 = load_bT(WhhT + k0, 256, n0);
      v16bf w1 = load_bT(WhhT + k0, 256, n0 + 16);
      acc0 = wmma_bf16(a, w0, acc0);
      acc1 = wmma_bf16(a, w1, acc1);
    }
#pragma unroll
    for (int r = 0; r < 8; ++r) {
      gls[(r + 8 * half) * 1024 + n0 + n] = acc0[r];
      gls[(r + 8 * half) * 1024 + n0 + 16 + n] = acc1[r];
    }
    __syncthreads();
#pragma unroll
    for (int e = 0; e < 4; ++e) {
      int id = e * 1024 + tid;
      int b = id >> 8, jj = id & 255;
      float gi = gls[b * 1024 + jj];
      float gf = gls[b * 1024 + 256 + jj];
      float gg = gls[b * 1024 + 512 + jj];
      float go = gls[b * 1024 + 768 + jj];
      float c = sigf(gf) * cls[b * 256 + jj] + sigf(gi) * tanh_fast(gg);
      float h = sigf(go) * tanh_fast(c);
      cls[b * 256 + jj] = c;
      hls[b * 256 + jj] = (__bf16)h;
      lo[((size_t)(b * S_ + t)) * 256 + jj] = h;
    }
    __syncthreads();
  }
}

// ---------------------------------------------------------------------------
// Flash attention, hd=32 (== one WMMA K-step). Q/K: (B,NH,S,32) bf16,
// V transposed: (B,NH,32,S) bf16 so P@V's B-operand is b128 loads.
// Block = 256 thr (8 waves) covers 128 query rows of one (b,h).
// window <= 0: full non-causal; window > 0: causal local (kj<=qi && kj>qi-w).
// V loads are issued before the softmax VALU section to overlap VMEM latency.
// ---------------------------------------------------------------------------
__global__ void flash_attn_kernel(const __bf16* __restrict__ Q,
                                  const __bf16* __restrict__ Km,
                                  const __bf16* __restrict__ Vt,
                                  float* __restrict__ O, int window) {
  __shared__ __align__(16) __bf16 Pl[8][16 * 32];
  int wid = threadIdx.x >> 5, lane = threadIdx.x & 31;
  int half = lane >> 4, n = lane & 15;
  int h = blockIdx.y, b = blockIdx.z;
  int q0b = blockIdx.x * 128;
  int q0 = q0b + wid * 16;
  size_t base = ((size_t)(b * NH_ + h)) * S_ * HD_;
  const __bf16* Qb = Q + base;
  const __bf16* Kb = Km + base;
  const __bf16* Vb = Vt + base;  // (32 x S) row-major
  v16bf qf = load_a(Qb, HD_, q0, 0, S_);
  v8f o0 = {}, o1 = {};
  float m[8], l[8];
#pragma unroll
  for (int r = 0; r < 8; ++r) { m[r] = -1e30f; l[r] = 0.f; }
  // Block-uniform key range so __syncthreads stays uniform across waves.
  int kstart = 0, kend = S_;
  if (window > 0) {
    kstart = q0b - (window - 1); if (kstart < 0) kstart = 0; kstart &= ~31;
    kend = q0b + 128;
  }
  const float scl = 0.17677669529663687f;  // 1/sqrt(32)
  for (int k0 = kstart; k0 < kend; k0 += 32) {
    v16bf kt0 = load_bT(Kb, HD_, k0);        // keys k0..k0+15
    v16bf kt1 = load_bT(Kb, HD_, k0 + 16);   // keys k0+16..k0+31
    v8f s0 = {}, s1 = {};
    s0 = wmma_bf16(qf, kt0, s0);
    s1 = wmma_bf16(qf, kt1, s1);
    // V fragment loads issued now; consumed after the softmax VALU below.
    v16bf v0 = load_bT(Vb + k0, S_, 0);    // V^T rows = hd cols 0..15
    v16bf v1 = load_bT(Vb + k0, S_, 16);   // hd cols 16..31
#pragma unroll
    for (int r = 0; r < 8; ++r) {
      int qi = q0 + r + 8 * half;
      float a0 = s0[r] * scl, a1 = s1[r] * scl;
      if (window > 0) {
        int kj0 = k0 + n, kj1 = k0 + 16 + n;
        if (!(kj0 <= qi && kj0 > qi - window)) a0 = -1e30f;
        if (!(kj1 <= qi && kj1 > qi - window)) a1 = -1e30f;
      }
      float mx = fmaxf(a0, a1);
#pragma unroll
      for (int off = 8; off; off >>= 1) mx = fmaxf(mx, __shfl_xor(mx, off, 32));
      float mn = fmaxf(m[r], mx);
      float al = __expf(m[r] - mn);
      float p0 = (a0 < -1e29f) ? 0.f : __expf(a0 - mn);
      float p1 = (a1 < -1e29f) ? 0.f : __expf(a1 - mn);
      float ps = p0 + p1;
#pragma unroll
      for (int off = 8; off; off >>= 1) ps += __shfl_xor(ps, off, 32);
      l[r] = l[r] * al + ps;
      m[r] = mn;
      int rr = r + 8 * half;
      Pl[wid][rr * 32 + n] = (__bf16)p0;
      Pl[wid][rr * 32 + 16 + n] = (__bf16)p1;
      o0[r] *= al;
      o1[r] *= al;
    }
    __syncthreads();
    v16bf pf = load_a(&Pl[wid][0], 32, 0, 0, 16);
    o0 = wmma_bf16(pf, v0, o0);
    o1 = wmma_bf16(pf, v1, o1);
    __syncthreads();
  }
#pragma unroll
  for (int r = 0; r < 8; ++r) {
    int row = q0 + r + 8 * half;
    float inv = 1.f / l[r];
    float* op = O + ((size_t)(b * S_) + row) * D_ + h * HD_;
    op[n] = o0[r] * inv;
    op[16 + n] = o1[r] * inv;
  }
}

// ---------------------------------------------------------------------------
// Fused embedding + fourier + rotary. One thread per (b,s,pair).
// ---------------------------------------------------------------------------
__global__ void embed_kernel(const int* __restrict__ locations,
                             const int* __restrict__ users,
                             const int* __restrict__ start_mins,
                             const int* __restrict__ weekdays,
                             const float* __restrict__ loc_emb,
                             const float* __restrict__ user_emb,
                             const float* __restrict__ wd_emb,
                             const float* __restrict__ freqs,
                             const float* __restrict__ rot_scale,
                             float* __restrict__ xf, __bf16* __restrict__ xb) {
  int id = blockIdx.x * 256 + threadIdx.x;  // B*S*128 total
  int i = id & 127;
  int s = (id >> 7) & 511;
  int b = id >> 16;
  int bs = b * S_ + s;
  int loc = locations[bs], usr = users[b * S_], wd = weekdays[bs];
  float hf = (float)((start_mins[bs] / 60) % 24);
  int d0 = 2 * i, d1 = 2 * i + 1;
  float f0 = (d0 < 128) ? __sinf(hf * freqs[d0]) : __cosf(hf * freqs[d0 - 128]);
  float f1 = (d1 < 128) ? __sinf(hf * freqs[d1]) : __cosf(hf * freqs[d1 - 128]);
  float e0 = loc_emb[(size_t)loc * D_ + d0] + user_emb[(size_t)usr * D_ + d0] +
             wd_emb[wd * D_ + d0] + f0;
  float e1 = loc_emb[(size_t)loc * D_ + d1] + user_emb[(size_t)usr * D_ + d1] +
             wd_emb[wd * D_ + d1] + f1;
  float ang = (hf / 24.f) * 6.2831853071795864f * rot_scale[i];
  float cc = __cosf(ang), ss = __sinf(ang);
  float y0 = e0 * cc - e1 * ss;
  float y1 = e0 * ss + e1 * cc;
  size_t o = (size_t)bs * D_;
  xf[o + d0] = y0; xf[o + d1] = y1;
  xb[o + d0] = (__bf16)y0; xb[o + d1] = (__bf16)y1;
}

// LayerNorm over D=256; 8 waves/block, one wave per row; optional residual,
// f32 out plus optional strided bf16 out (for building concat inputs).
__global__ void layernorm_kernel(const float* __restrict__ A,
                                 const float* __restrict__ R,
                                 const float* __restrict__ g,
                                 const float* __restrict__ bt,
                                 float* __restrict__ Of,
                                 __bf16* __restrict__ Ob, int obs, int obo,
                                 int rows) {
  int wid = threadIdx.x >> 5, lane = threadIdx.x & 31;
  int row = blockIdx.x * 8 + wid;
  if (row >= rows) return;
  const float* a = A + (size_t)row * D_;
  const float* r = R ? R + (size_t)row * D_ : nullptr;
  float v[8], s = 0.f;
#pragma unroll
  for (int j = 0; j < 8; ++j) {
    float x = a[lane + j * 32];
    if (r) x += r[lane + j * 32];
    v[j] = x; s += x;
  }
#pragma unroll
  for (int off = 16; off; off >>= 1) s += __shfl_xor(s, off, 32);
  float mean = s * (1.f / 256.f), vs = 0.f;
#pragma unroll
  for (int j = 0; j < 8; ++j) { float d = v[j] - mean; vs += d * d; }
#pragma unroll
  for (int off = 16; off; off >>= 1) vs += __shfl_xor(vs, off, 32);
  float inv = rsqrtf(vs * (1.f / 256.f) + 1e-5f);
#pragma unroll
  for (int j = 0; j < 8; ++j) {
    int col = lane + j * 32;
    float y = (v[j] - mean) * inv * g[col] + bt[col];
    Of[(size_t)row * D_ + col] = y;
    if (Ob) Ob[(size_t)row * obs + obo + col] = (__bf16)y;
  }
}

__global__ void convert_bf16_kernel(const float* __restrict__ s,
                                    __bf16* __restrict__ d, size_t n) {
  size_t i = (size_t)blockIdx.x * 256 + threadIdx.x;
  if (i < n) d[i] = (__bf16)s[i];
}

// W (K x N) f32 -> WT (N x K) bf16
__global__ void convert_T_kernel(const float* __restrict__ s,
                                 __bf16* __restrict__ d, int K, int N) {
  int id = blockIdx.x * 256 + threadIdx.x;
  if (id >= K * N) return;
  int k = id / N, n = id % N;
  d[(size_t)n * K + k] = (__bf16)s[id];
}

// (B,S,3D) f32 -> q,k: (B,NH,S,32) bf16 ; v: transposed (B,NH,32,S) bf16
__global__ void qkv_pack_kernel(const float* __restrict__ qkv,
                                __bf16* __restrict__ q, __bf16* __restrict__ k,
                                __bf16* __restrict__ vt) {
  int id = blockIdx.x * 256 + threadIdx.x;  // B*NH*S*32
  int c = id & 31, s = (id >> 5) & 511, h = (id >> 14) & 7, b = id >> 17;
  size_t src = ((size_t)(b * S_ + s)) * (3 * D_) + h * HD_ + c;
  q[id] = (__bf16)qkv[src];
  k[id] = (__bf16)qkv[src + D_];
  vt[(((size_t)(b * NH_ + h)) * HD_ + c) * S_ + s] = (__bf16)qkv[src + 2 * D_];
}

__global__ void gate_combine_kernel(const float* __restrict__ gw,
                                    const float* __restrict__ lo,
                                    const float* __restrict__ ao,
                                    float* __restrict__ of,
                                    __bf16* __restrict__ ob, size_t n) {
  size_t i = (size_t)blockIdx.x * 256 + threadIdx.x;
  if (i >= n) return;
  float g = gw[i];
  float v = g * lo[i] + (1.f - g) * ao[i];
  of[i] = v; ob[i] = (__bf16)v;
}

__global__ void gather_last_kernel(const float* __restrict__ x,
                                   __bf16* __restrict__ fr) {
  int id = blockIdx.x * 256 + threadIdx.x;  // 16*256
  int b = id >> 8, d = id & 255;
  fr[id] = (__bf16)x[((size_t)(b * S_ + S_ - 1)) * D_ + d];
}

// Row softmax, 64 cols, 1/temp scale; in-place f32 + TRANSPOSED bf16 mirror
// (AbT: 64 x rows) so the downstream probs@assign GEMM gets b128 B-operands.
__global__ void softmax_rows64_kernel(float* __restrict__ A,
                                      __bf16* __restrict__ AbT, int rows,
                                      const float* __restrict__ temp) {
  int wid = threadIdx.x >> 5, lane = threadIdx.x & 31;
  int row = blockIdx.x * 8 + wid;
  if (row >= rows) return;
  float it = 1.f / temp[0];
  float* a = A + (size_t)row * NC_;
  float x0 = a[lane] * it, x1 = a[lane + 32] * it;
  float mx = fmaxf(x0, x1);
#pragma unroll
  for (int off = 16; off; off >>= 1) mx = fmaxf(mx, __shfl_xor(mx, off, 32));
  float e0 = __expf(x0 - mx), e1 = __expf(x1 - mx);
  float s = e0 + e1;
#pragma unroll
  for (int off = 16; off; off >>= 1) s += __shfl_xor(s, off, 32);
  float inv = 1.f / s;
  a[lane] = e0 * inv; a[lane + 32] = e1 * inv;
  AbT[(size_t)lane * rows + row] = (__bf16)(e0 * inv);
  AbT[(size_t)(lane + 32) * rows + row] = (__bf16)(e1 * inv);
}

// Softmax over 40000 cols; block (1024 thr) per batch row. Leaves logits
// untouched, writes bf16 probs.
__global__ void softmax_bigrow_kernel(const float* __restrict__ logits,
                                      __bf16* __restrict__ probs, int cols) {
  __shared__ float red[32];
  int tid = threadIdx.x, wid = tid >> 5, lane = tid & 31;
  const float* a = logits + (size_t)blockIdx.x * cols;
  float mx = -1e30f;
  for (int c = tid; c < cols; c += 1024) mx = fmaxf(mx, a[c]);
#pragma unroll
  for (int off = 16; off; off >>= 1) mx = fmaxf(mx, __shfl_xor(mx, off, 32));
  if (lane == 0) red[wid] = mx;
  __syncthreads();
  if (wid == 0) {
    float m2 = red[lane];
#pragma unroll
    for (int off = 16; off; off >>= 1) m2 = fmaxf(m2, __shfl_xor(m2, off, 32));
    if (lane == 0) red[0] = m2;
  }
  __syncthreads();
  mx = red[0];
  __syncthreads();
  float s = 0.f;
  for (int c = tid; c < cols; c += 1024) s += __expf(a[c] - mx);
#pragma unroll
  for (int off = 16; off; off >>= 1) s += __shfl_xor(s, off, 32);
  if (lane == 0) red[wid] = s;
  __syncthreads();
  if (wid == 0) {
    float s2 = red[lane];
#pragma unroll
    for (int off = 16; off; off >>= 1) s2 += __shfl_xor(s2, off, 32);
    if (lane == 0) red[0] = s2;
  }
  __syncthreads();
  float inv = 1.f / red[0];
  __bf16* p = probs + (size_t)blockIdx.x * cols;
  for (int c = tid; c < cols; c += 1024) p[c] = (__bf16)(__expf(a[c] - mx) * inv);
}

__global__ void colsum_kernel(const float* __restrict__ assign,
                              float* __restrict__ cs, int rows) {
  __shared__ float red[8];
  int col = blockIdx.x, tid = threadIdx.x, wid = tid >> 5, lane = tid & 31;
  float s = 0.f;
  for (int r = tid; r < rows; r += 256) s += assign[(size_t)r * NC_ + col];
#pragma unroll
  for (int off = 16; off; off >>= 1) s += __shfl_xor(s, off, 32);
  if (lane == 0) red[wid] = s;
  __syncthreads();
  if (tid == 0) {
    float t = 0.f;
    for (int w = 0; w < 8; ++w) t += red[w];
    cs[col] = t;
  }
}

// cluster_targets = softmax(one_hot) @ assign, computed analytically:
// softmax row = {e/(e+NLOC-1) at target, 1/(e+NLOC-1) elsewhere}.
__global__ void cluster_targets_kernel(const float* __restrict__ assign,
                                       const float* __restrict__ cs,
                                       const int* __restrict__ target,
                                       float* __restrict__ out) {
  int tid = threadIdx.x;  // 1024 == 16*64
  int b = tid >> 6, j = tid & 63;
  const float e = 2.7182818284590452f;
  float denom = e + (float)(NLOC_ - 1);
  float pt = e / denom, po = 1.f / denom;
  out[b * NC_ + j] = po * cs[j] + (pt - po) * assign[(size_t)target[b] * NC_ + j];
}

// ---------------------------------------------------------------------------
extern "C" void kernel_launch(void* const* d_in, const int* in_sizes, int n_in,
                              void* d_out, int out_size, void* d_ws,
                              size_t ws_size, hipStream_t stream) {
  (void)in_sizes; (void)n_in; (void)out_size; (void)ws_size;
  // ---- inputs (setup_inputs dict order, params flattened in insertion order)
  const int* locations = (const int*)d_in[0];
  const int* users = (const int*)d_in[1];
  const int* start_mins = (const int*)d_in[2];
  const int* weekdays = (const int*)d_in[3];
  const int* target = (const int*)d_in[4];
  const float* loc_emb = (const float*)d_in[5];
  const float* user_emb = (const float*)d_in[6];
  const float* wd_emb = (const float*)d_in[7];
  const float* freqs = (const float*)d_in[8];
  const float* rot_scale = (const float*)d_in[9];
  const float* lstm_Wih = (const float*)d_in[10];
  const float* lstm_Whh = (const float*)d_in[11];
  const float* lstm_b = (const float*)d_in[12];
  const float* attn_Wqkv = (const float*)d_in[13];
  const float* attn_bq = (const float*)d_in[14];
  const float* attn_Wo = (const float*)d_in[15];
  const float* attn_bo = (const float*)d_in[16];
  const float* ffn_W1 = (const float*)d_in[17];
  const float* ffn_b1 = (const float*)d_in[18];
  const float* ffn_W2 = (const float*)d_in[19];
  const float* ffn_b2 = (const float*)d_in[20];
  const float* ln_g = (const float*)d_in[21];
  const float* ln_b = (const float*)d_in[22];
  const float* gate_W = (const float*)d_in[23];
  const float* gate_b = (const float*)d_in[24];
  const float* ms_Wqkv = (const float*)d_in[25];
  const float* ms_bq = (const float*)d_in[26];
  const float* ms_Wo = (const float*)d_in[27];
  const float* ms_bo = (const float*)d_in[28];
  const float* fusion_W = (const float*)d_in[29];
  const float* fusion_b = (const float*)d_in[30];
  const float* clh_W = (const float*)d_in[31];
  const float* clh_b = (const float*)d_in[32];
  const float* loch_W = (const float*)d_in[33];
  const float* loch_b = (const float*)d_in[34];
  const float* centers = (const float*)d_in[35];
  const float* clu_loc_emb = (const float*)d_in[36];
  const float* temp = (const float*)d_in[37];
  float* out = (float*)d_out;

  // ---- workspace bump allocator
  size_t cur = 0;
  auto alloc = [&](size_t bytes) -> void* {
    void* p = (char*)d_ws + cur;
    cur += (bytes + 255) & ~(size_t)255;
    return p;
  };
  // bf16 TRANSPOSED weights (N x K), L2-resident after first pass
  __bf16* wihT = (__bf16*)alloc((size_t)L_ * D_ * 4 * D_ * 2);
  __bf16* whhT = (__bf16*)alloc((size_t)L_ * D_ * 4 * D_ * 2);
  __bf16* wqkvT = (__bf16*)alloc((size_t)L_ * D_ * 3 * D_ * 2);
  __bf16* woT = (__bf16*)alloc((size_t)L_ * D_ * D_ * 2);
  __bf16* w1T = (__bf16*)alloc((size_t)L_ * D_ * FF_ * 2);
  __bf16* w2T = (__bf16*)alloc((size_t)L_ * FF_ * D_ * 2);
  __bf16* gateT = (__bf16*)alloc((size_t)L_ * 2 * D_ * D_ * 2);
  __bf16* msqkvT = (__bf16*)alloc((size_t)3 * D_ * 3 * D_ * 2);
  __bf16* mswoT = (__bf16*)alloc((size_t)3 * D_ * D_ * 2);
  __bf16* fusT = (__bf16*)alloc((size_t)3 * D_ * D_ * 2);
  __bf16* clhT = (__bf16*)alloc((size_t)D_ * NC_ * 2);
  __bf16* lochT = (__bf16*)alloc((size_t)D_ * NLOC_ * 2);
  __bf16* centNK = (__bf16*)alloc((size_t)D_ * NC_ * 2);   // centers is already N x K
  __bf16* clu_bf = (__bf16*)alloc((size_t)NLOC_ * D_ * 2);
  // activations
  float* x_f = (float*)alloc((size_t)BS_ * D_ * 4);
  __bf16* x_bf = (__bf16*)alloc((size_t)BS_ * D_ * 2);
  float* xg_f = (float*)alloc((size_t)BS_ * 4 * D_ * 4);
  float* lo_f = (float*)alloc((size_t)BS_ * D_ * 4);
  float* lnl_f = (float*)alloc((size_t)BS_ * D_ * 4);
  float* lna_f = (float*)alloc((size_t)BS_ * D_ * 4);
  float* qkv_f = (float*)alloc((size_t)BS_ * 3 * D_ * 4);
  __bf16* qb = (__bf16*)alloc((size_t)BS_ * D_ * 2);
  __bf16* kb = (__bf16*)alloc((size_t)BS_ * D_ * 2);
  __bf16* vtb = (__bf16*)alloc((size_t)BS_ * D_ * 2);
  float* attno_f = (float*)alloc((size_t)BS_ * D_ * 4);
  __bf16* attno_bf = (__bf16*)alloc((size_t)BS_ * D_ * 2);
  float* proj_f = (float*)alloc((size_t)BS_ * D_ * 4);
  __bf16* concat_bf = (__bf16*)alloc((size_t)BS_ * 2 * D_ * 2);
  float* gwv_f = (float*)alloc((size_t)BS_ * D_ * 4);
  float* gated_f = (float*)alloc((size_t)BS_ * D_ * 4);
  __bf16* gated_bf = (__bf16*)alloc((size_t)BS_ * D_ * 2);
  float* ffn1_f = (float*)alloc((size_t)BS_ * FF_ * 4);
  __bf16* ffn1_bf = (__bf16*)alloc((size_t)BS_ * FF_ * 2);
  float* ffn2_f = (float*)alloc((size_t)BS_ * D_ * 4);
  __bf16* mscat_bf = (__bf16*)alloc((size_t)BS_ * 3 * D_ * 2);
  float* fusout_f = (float*)alloc((size_t)BS_ * D_ * 4);
  __bf16* fr_bf = (__bf16*)alloc((size_t)B_ * D_ * 2);
  float* assign_f = (float*)alloc((size_t)NLOC_ * NC_ * 4);
  __bf16* assignT_bf = (__bf16*)alloc((size_t)NLOC_ * NC_ * 2);  // 64 x 40000
  __bf16* probs_bf = (__bf16*)alloc((size_t)B_ * NLOC_ * 2);
  float* colsum_f = (float*)alloc(NC_ * 4);

  auto conv = [&](const float* s, __bf16* d, size_t n) {
    convert_bf16_kernel<<<(unsigned)((n + 255) / 256), 256, 0, stream>>>(s, d, n);
  };
  auto convT = [&](const float* s, __bf16* d, int K, int N) {
    convert_T_kernel<<<(K * N + 255) / 256, 256, 0, stream>>>(s, d, K, N);
  };
  auto gemm = [&](const __bf16* A, int lda, const __bf16* WT, const float* bias,
                  float* C, int ldc, __bf16* Cb, int ldcb, int M, int N, int K,
                  int act) {
    dim3 g(N / 64, (M + 31) / 32);
    gemm_bf16_kernel<<<g, 128, 0, stream>>>(A, lda, WT, bias, C, ldc, Cb, ldcb,
                                            M, N, K, act);
  };

  // ---- one-time (per launch) transposed weight converts
  for (int l = 0; l < L_; ++l) {
    convT(lstm_Wih + (size_t)l * D_ * 4 * D_, wihT + (size_t)l * D_ * 4 * D_, D_, 4 * D_);
    convT(lstm_Whh + (size_t)l * D_ * 4 * D_, whhT + (size_t)l * D_ * 4 * D_, D_, 4 * D_);
    convT(attn_Wqkv + (size_t)l * D_ * 3 * D_, wqkvT + (size_t)l * D_ * 3 * D_, D_, 3 * D_);
    convT(attn_Wo + (size_t)l * D_ * D_, woT + (size_t)l * D_ * D_, D_, D_);
    convT(ffn_W1 + (size_t)l * D_ * FF_, w1T + (size_t)l * D_ * FF_, D_, FF_);
    convT(ffn_W2 + (size_t)l * FF_ * D_, w2T + (size_t)l * FF_ * D_, FF_, D_);
    convT(gate_W + (size_t)l * 2 * D_ * D_, gateT + (size_t)l * 2 * D_ * D_, 2 * D_, D_);
  }
  for (int si = 0; si < 3; ++si) {
    convT(ms_Wqkv + (size_t)si * D_ * 3 * D_, msqkvT + (size_t)si * D_ * 3 * D_, D_, 3 * D_);
    convT(ms_Wo + (size_t)si * D_ * D_, mswoT + (size_t)si * D_ * D_, D_, D_);
  }
  convT(fusion_W, fusT, 3 * D_, D_);
  convT(clh_W, clhT, D_, NC_);
  convT(loch_W, lochT, D_, NLOC_);
  conv(centers, centNK, (size_t)NC_ * D_);  // centers (NC,D) == (N,K) already
  conv(clu_loc_emb, clu_bf, (size_t)NLOC_ * D_);

  // ---- embedding + rotary
  embed_kernel<<<(B_ * S_ * 128) / 256, 256, 0, stream>>>(
      locations, users, start_mins, weekdays, loc_emb, user_emb, wd_emb, freqs,
      rot_scale, x_f, x_bf);

  const int lnBlocks = BS_ / 8;
  // ---- transformer/LSTM hybrid layers
  for (int l = 0; l < L_; ++l) {
    // LSTM input gates: xg = x @ Wih + b
    gemm(x_bf, D_, wihT + (size_t)l * D_ * 4 * D_, lstm_b + (size_t)l * 4 * D_,
         xg_f, 4 * D_, nullptr, 0, BS_, 4 * D_, D_, 0);
    // recurrence (single persistent workgroup)
    lstm_rec_kernel<<<1, 1024, 0, stream>>>(
        xg_f, whhT + (size_t)l * D_ * 4 * D_, lo_f);
    // lo = LN(lo + x)  (bf16 into concat[:,0:256])
    layernorm_kernel<<<lnBlocks, 256, 0, stream>>>(
        lo_f, x_f, ln_g + (size_t)(l * 3 + 0) * D_, ln_b + (size_t)(l * 3 + 0) * D_,
        lnl_f, concat_bf, 2 * D_, 0, BS_);
    // attention branch
    gemm(x_bf, D_, wqkvT + (size_t)l * D_ * 3 * D_, attn_bq + (size_t)l * 3 * D_,
         qkv_f, 3 * D_, nullptr, 0, BS_, 3 * D_, D_, 0);
    qkv_pack_kernel<<<(B_ * NH_ * S_ * HD_) / 256, 256, 0, stream>>>(qkv_f, qb, kb, vtb);
    flash_attn_kernel<<<dim3(S_ / 128, NH_, B_), 256, 0, stream>>>(qb, kb, vtb,
                                                                   attno_f, -1);
    conv(attno_f, attno_bf, (size_t)BS_ * D_);
    gemm(attno_bf, D_, woT + (size_t)l * D_ * D_, attn_bo + (size_t)l * D_,
         proj_f, D_, nullptr, 0, BS_, D_, D_, 0);
    // ao = LN(proj + x)  (bf16 into concat[:,256:512])
    layernorm_kernel<<<lnBlocks, 256, 0, stream>>>(
        proj_f, x_f, ln_g + (size_t)(l * 3 + 1) * D_, ln_b + (size_t)(l * 3 + 1) * D_,
        lna_f, concat_bf, 2 * D_, D_, BS_);
    // gate = sigmoid(concat @ gate_W + b)
    gemm(concat_bf, 2 * D_, gateT + (size_t)l * 2 * D_ * D_,
         gate_b + (size_t)l * D_, gwv_f, D_, nullptr, 0, BS_, D_, 2 * D_, 1);
    gate_combine_kernel<<<(BS_ * D_) / 256, 256, 0, stream>>>(
        gwv_f, lnl_f, lna_f, gated_f, gated_bf, (size_t)BS_ * D_);
    // FFN with exact gelu
    gemm(gated_bf, D_, w1T + (size_t)l * D_ * FF_, ffn_b1 + (size_t)l * FF_,
         ffn1_f, FF_, ffn1_bf, FF_, BS_, FF_, D_, 2);
    gemm(ffn1_bf, FF_, w2T + (size_t)l * FF_ * D_, ffn_b2 + (size_t)l * D_,
         ffn2_f, D_, nullptr, 0, BS_, D_, FF_, 0);
    // x = LN(ffn2 + gated)
    layernorm_kernel<<<lnBlocks, 256, 0, stream>>>(
        ffn2_f, gated_f, ln_g + (size_t)(l * 3 + 2) * D_,
        ln_b + (size_t)(l * 3 + 2) * D_, x_f, x_bf, D_, 0, BS_);
  }

  // ---- multi-scale attention heads (windows 5, 10, full)
  const int windows[3] = {5, 10, -1};
  for (int si = 0; si < 3; ++si) {
    gemm(x_bf, D_, msqkvT + (size_t)si * D_ * 3 * D_, ms_bq + (size_t)si * 3 * D_,
         qkv_f, 3 * D_, nullptr, 0, BS_, 3 * D_, D_, 0);
    qkv_pack_kernel<<<(B_ * NH_ * S_ * HD_) / 256, 256, 0, stream>>>(qkv_f, qb, kb, vtb);
    flash_attn_kernel<<<dim3(S_ / 128, NH_, B_), 256, 0, stream>>>(
        qb, kb, vtb, attno_f, windows[si]);
    conv(attno_f, attno_bf, (size_t)BS_ * D_);
    // Wo projection; bf16 mirror lands in the (B*S, 3D) concat buffer.
    gemm(attno_bf, D_, mswoT + (size_t)si * D_ * D_, ms_bo + (size_t)si * D_,
         proj_f, D_, mscat_bf + (size_t)si * D_, 3 * D_, BS_, D_, D_, 0);
  }
  // fusion
  gemm(mscat_bf, 3 * D_, fusT, fusion_b, fusout_f, D_, nullptr, 0,
       BS_, D_, 3 * D_, 0);
  gather_last_kernel<<<(B_ * D_) / 256, 256, 0, stream>>>(fusout_f, fr_bf);

  // ---- heads
  const size_t O1 = (size_t)B_ * NLOC_;          // cluster_logits
  const size_t O2 = O1 + (size_t)B_ * NC_;       // cluster_probs
  const size_t O3 = O2 + (size_t)B_ * NC_;       // cluster_targets
  // location_logits -> d_out[0 : 16*40000]
  gemm(fr_bf, D_, lochT, loch_b, out, NLOC_, nullptr, 0, B_, NLOC_, D_, 0);
  // cluster_logits
  gemm(fr_bf, D_, clhT, clh_b, out + O1, NC_, nullptr, 0, B_, NC_, D_, 0);
  // assign = softmax(clu_loc_emb @ centers.T / temp)
  gemm(clu_bf, D_, centNK, nullptr, assign_f, NC_, nullptr, 0, NLOC_, NC_, D_, 0);
  softmax_rows64_kernel<<<NLOC_ / 8, 256, 0, stream>>>(assign_f, assignT_bf,
                                                       NLOC_, temp);
  // cluster_probs = softmax(location_logits) @ assign
  softmax_bigrow_kernel<<<B_, 1024, 0, stream>>>(out, probs_bf, NLOC_);
  gemm(probs_bf, NLOC_, assignT_bf, nullptr, out + O2, NC_, nullptr, 0,
       B_, NC_, NLOC_, 0);
  // cluster_targets (analytic softmax of one-hot)
  colsum_kernel<<<NC_, 256, 0, stream>>>(assign_f, colsum_f, NLOC_);
  cluster_targets_kernel<<<1, 1024, 0, stream>>>(assign_f, colsum_f, target,
                                                 out + O3);
}
